// AutoEncoder_16741782520503
// MI455X (gfx1250) — matrix operand order repaired and checked by reading, not compiled, not run on hardware
//
#include <hip/hip_runtime.h>
#include <hip/hip_bf16.h>

typedef __attribute__((ext_vector_type(16))) __bf16 v16bf;
typedef __attribute__((ext_vector_type(8)))  float  v8f;
typedef __attribute__((ext_vector_type(4)))  unsigned int u32x4;
typedef __attribute__((ext_vector_type(8)))  int i32x8;
typedef __attribute__((ext_vector_type(4)))  int i32x4;

#define XSZ   128
#define NPIX  16384
#define BATCH 64
#define HD    1024
#define NPTS  65536

#define GEMM_HAS_TDM (__has_builtin(__builtin_amdgcn_tensor_load_to_lds) && \
                      __has_builtin(__builtin_amdgcn_s_wait_tensorcnt))

__device__ inline __bf16 f2bf(float x) { return (__bf16)x; }

// ---------------------------------------------------------------------------
// Tensor Data Mover: DMA a 64-row x 32-col f32 tile (row stride = K elements)
// from global memory into LDS at byte offset lds_off. D# packing per CDNA5
// ISA 8.3/8.4: group0 {count=1, lds_addr, global_addr, type=2},
// group1 {data_size=4B, tensor_dim0=K, tensor_dim1=64, tile_dim0=32,
//         tile_dim1=64, tensor_dim0_stride=K}.
// ---------------------------------------------------------------------------
__device__ inline void tdm_load_slab(const float* gsrc, unsigned lds_off, int K) {
#if GEMM_HAS_TDM
  const unsigned long long ga = (unsigned long long)(uintptr_t)gsrc;
  u32x4 g0 = {};
  g0[0] = 1u;                                                  // count=1, is_restore=0
  g0[1] = lds_off;                                             // lds_addr  [63:32]
  g0[2] = (unsigned)ga;                                        // global_addr[31:0]
  g0[3] = (unsigned)((ga >> 32) & 0x01FFFFFFu) | 0x80000000u;  // addr[56:32], type=2
  i32x8 g1 = {};
  g1[0] = (int)(2u << 16);                                     // wg_mask=0, data_size=4B
  g1[1] = (int)(((unsigned)K & 0xFFFFu) << 16);                // tensor_dim0[15:0] @63:48
  g1[2] = (int)((((unsigned)K >> 16) & 0xFFFFu) | (64u << 16));// dim0[31:16], dim1=64
  g1[3] = (int)(32u << 16);                                    // dim1_hi=0, tile_dim0=32
  g1[4] = 64;                                                  // tile_dim1=64, tile_dim2=0
  g1[5] = K;                                                   // tensor_dim0_stride[31:0]
  g1[6] = 0;                                                   // stride0_hi, stride1_lo
  g1[7] = 0;
  i32x4 z4 = {};
#if __clang_major__ >= 23
  i32x8 z8 = {};
  __builtin_amdgcn_tensor_load_to_lds(g0, g1, z4, z4, z8, 0);
#else
  __builtin_amdgcn_tensor_load_to_lds(g0, g1, z4, z4, 0);
#endif
#else
  (void)gsrc; (void)lds_off; (void)K;
#endif
}

// ---------------------------------------------------------------------------
// Per-image standardization: (x - mean) / max(std, 1/sqrt(n))
// ---------------------------------------------------------------------------
__global__ void k_standardize(const float* __restrict__ img, float* __restrict__ out) {
  __shared__ float s_sum[256], s_sq[256];
  const int b = blockIdx.x, tid = threadIdx.x;
  const float* x = img + (size_t)b * NPIX;
  float sum = 0.f, sq = 0.f;
  for (int i = tid; i < NPIX; i += 256) { float v = x[i]; sum += v; sq += v * v; }
  s_sum[tid] = sum; s_sq[tid] = sq; __syncthreads();
  for (int s = 128; s > 0; s >>= 1) {
    if (tid < s) { s_sum[tid] += s_sum[tid + s]; s_sq[tid] += s_sq[tid + s]; }
    __syncthreads();
  }
  const float mean = s_sum[0] * (1.0f / NPIX);
  const float var  = s_sq[0] * (1.0f / NPIX) - mean * mean;
  const float sd   = sqrtf(fmaxf(var, 0.f));
  const float inv  = 1.0f / fmaxf(sd, 1.0f / 128.0f);
  for (int i = tid; i < NPIX; i += 256) out[(size_t)b * NPIX + i] = (x[i] - mean) * inv;
}

// ---------------------------------------------------------------------------
// C[64 x N] = relu(A[64 x K] @ W[K x N] + bias)  (optional batchnorm after relu)
// One wave per block; wave owns one 16-col tile and all four 16-row tiles.
// A-slab (64x32 f32) is TDM-DMA'd into LDS, double-buffered, synchronized
// with s_wait_tensorcnt. bf16 WMMA, f32 accumulate.
// ---------------------------------------------------------------------------
__global__ __launch_bounds__(32)
void k_gemm_relu(const float* __restrict__ A, const float* __restrict__ W,
                 const float* __restrict__ bias, const float* __restrict__ bn,
                 float* __restrict__ C, int K, int N) {
  __shared__ float Asf[2][64 * 32];
  const int lane = threadIdx.x;
  const int ncol = blockIdx.x * 16 + (lane & 15);
  const int hi   = lane >> 4;                 // 0 or 1: which half-wave
  v8f acc[4] = {};

#if GEMM_HAS_TDM
  tdm_load_slab(A, (unsigned)(uintptr_t)&Asf[0][0], K);
#endif

  int cur = 0;
  for (int k0 = 0; k0 < K; k0 += 32, cur ^= 1) {
#if GEMM_HAS_TDM
    if (k0 + 32 < K)
      tdm_load_slab(A + (k0 + 32), (unsigned)(uintptr_t)&Asf[cur ^ 1][0], K);
#else
    for (int i = lane; i < 64 * 32; i += 32)
      Asf[cur][i] = A[(size_t)(i >> 5) * K + k0 + (i & 31)];
    __syncthreads();
#endif

    // B fragment (32x16 bf16): lane holds N=ncol, K = hi*16 + e  (from global)
    v16bf bf;
    #pragma unroll
    for (int e = 0; e < 16; ++e)
      bf[e] = f2bf(W[(size_t)(k0 + hi * 16 + e) * N + ncol]);

    if (k0 + 64 < K)
      __builtin_prefetch(&W[(size_t)(k0 + 64) * N + ncol], 0, 1);

#if GEMM_HAS_TDM
    if (k0 + 32 < K) __builtin_amdgcn_s_wait_tensorcnt((short)1);
    else             __builtin_amdgcn_s_wait_tensorcnt((short)0);
    __asm__ volatile("" ::: "memory");
#endif

    // A fragments (16x32 bf16) for 4 M-tiles; layout matches B:
    // lanes 0-15: e0..15 -> K=0..15
    // lanes16-31: e0..15 -> K=16..31
    const float* slab = Asf[cur];
    const int kb = hi * 16;
    #pragma unroll
    for (int mt = 0; mt < 4; ++mt) {
      const float* arow = &slab[(mt * 16 + (lane & 15)) * 32];
      v16bf af;
      #pragma unroll
      for (int e = 0; e < 8; ++e) af[e]     = f2bf(arow[kb + e]);
      #pragma unroll
      for (int e = 0; e < 8; ++e) af[8 + e] = f2bf(arow[kb + 8 + e]);
      acc[mt] = __builtin_amdgcn_wmma_f32_16x16x32_bf16(
          false, af, false, bf, (short)0, acc[mt], false, false);
    }
#if !GEMM_HAS_TDM
    __syncthreads();
#endif
  }

  // epilogue: bias + relu (+ batchnorm)
  const float bv = bias[ncol];
  float g = 1.f, beta = 0.f, mu = 0.f, iv = 1.f;
  if (bn) {
    g    = bn[ncol];
    beta = bn[HD + ncol];
    mu   = bn[2 * HD + ncol];
    iv   = rsqrtf(bn[3 * HD + ncol] + 0.001f);
  }
  #pragma unroll
  for (int mt = 0; mt < 4; ++mt) {
    #pragma unroll
    for (int r = 0; r < 8; ++r) {
      const int m = mt * 16 + hi * 8 + r;      // C layout: vgpr r -> M = r + 8*hi
      float v = acc[mt][r] + bv;
      v = fmaxf(v, 0.f);
      if (bn) v = (v - mu) * g * iv + beta;
      C[(size_t)m * N + ncol] = v;
    }
  }
}

// ---------------------------------------------------------------------------
// Heads: algn = y@W_algn+b, shf = z@W_shift+b, reparam, Euler rows.
// One block per batch element; outputs 8 floats: row0(3), row1(3), zs(2).
// ---------------------------------------------------------------------------
__global__ void k_heads(const float* __restrict__ y, const float* __restrict__ z,
                        const float* __restrict__ W_algn, const float* __restrict__ b_algn,
                        const float* __restrict__ W_shift, const float* __restrict__ b_shift,
                        const float* __restrict__ eps_a, const float* __restrict__ eps_s,
                        const float* __restrict__ rotb, const float* __restrict__ tiltb,
                        const float* __restrict__ psib, float* __restrict__ params) {
  __shared__ float red[256];
  __shared__ float outv[10];
  const int b = blockIdx.x, tid = threadIdx.x;
  float acc[10];
  #pragma unroll
  for (int j = 0; j < 10; ++j) acc[j] = 0.f;
  for (int k = tid; k < HD; k += 256) {
    const float yv = y[(size_t)b * HD + k], zv = z[(size_t)b * HD + k];
    #pragma unroll
    for (int j = 0; j < 6; ++j) acc[j] += yv * W_algn[k * 6 + j];
    #pragma unroll
    for (int j = 0; j < 4; ++j) acc[6 + j] += zv * W_shift[k * 4 + j];
  }
  for (int j = 0; j < 10; ++j) {
    red[tid] = acc[j]; __syncthreads();
    for (int s = 128; s > 0; s >>= 1) {
      if (tid < s) red[tid] += red[tid + s];
      __syncthreads();
    }
    if (tid == 0) outv[j] = red[0];
    __syncthreads();
  }
  if (tid == 0) {
    float algn[6], shf[4];
    for (int j = 0; j < 6; ++j) algn[j] = outv[j] + b_algn[j];
    for (int j = 0; j < 4; ++j) shf[j]  = outv[6 + j] + b_shift[j];
    float za[3], zs[2];
    for (int i = 0; i < 3; ++i) za[i] = eps_a[b * 3 + i] * expf(0.5f * algn[3 + i]) + algn[i];
    for (int i = 0; i < 2; ++i) zs[i] = eps_s[b * 2 + i] * expf(0.5f * shf[2 + i]) + shf[i];
    const float rot = rotb[b] + za[0], tilt = tiltb[b] + za[1], psi = psib[b] + za[2];
    float sa, ca, sb, cb, sg, cg;
    __sincosf(rot,  &sa, &ca);
    __sincosf(tilt, &sb, &cb);
    __sincosf(psi,  &sg, &cg);
    const float cc = cb * ca, cs = cb * sa;
    float* pp = params + b * 8;
    pp[0] =  cg * cc - sg * sa;  pp[1] =  cg * cs + sg * ca;  pp[2] = -cg * sb;
    pp[3] = -sg * cc - cg * sa;  pp[4] = -sg * cs + cg * ca;  pp[5] =  sg * sb;
    pp[6] = zs[0];               pp[7] = zs[1];
  }
}

// ---------------------------------------------------------------------------
// In-LDS radix-2 FFTs over a 128x128 complex grid.
// dir = -1: forward, +1: inverse (unscaled).
// ---------------------------------------------------------------------------
__device__ void fft128_rows(float2* c, float dir, int tid, int nth) {
  __syncthreads();
  for (int idx = tid; idx < NPIX; idx += nth) {
    const int row = idx >> 7, col = idx & 127;
    const int rc = __brev((unsigned)col) >> 25;
    if (col < rc) {
      float2 t = c[(row << 7) | col];
      c[(row << 7) | col] = c[(row << 7) | rc];
      c[(row << 7) | rc] = t;
    }
  }
  __syncthreads();
  for (int s = 1; s <= 7; ++s) {
    const int half = 1 << (s - 1);
    for (int idx = tid; idx < 8192; idx += nth) {
      const int row = idx >> 6, j = idx & 63;
      const int grp = j >> (s - 1), pos = j & (half - 1);
      const int i0 = (row << 7) + (grp << s) + pos;
      const int i1 = i0 + half;
      const float ang = dir * 3.14159265358979f * (float)pos / (float)half;
      float sn, cs;
      __sincosf(ang, &sn, &cs);
      const float2 u = c[i0], v = c[i1];
      const float tr = cs * v.x - sn * v.y;
      const float ti = cs * v.y + sn * v.x;
      c[i0] = make_float2(u.x + tr, u.y + ti);
      c[i1] = make_float2(u.x - tr, u.y - ti);
    }
    __syncthreads();
  }
}

__device__ void fft128_cols(float2* c, float dir, int tid, int nth) {
  __syncthreads();
  for (int idx = tid; idx < NPIX; idx += nth) {
    const int row = idx >> 7, col = idx & 127;
    const int rr = __brev((unsigned)row) >> 25;
    if (row < rr) {
      float2 t = c[(row << 7) | col];
      c[(row << 7) | col] = c[(rr << 7) | col];
      c[(rr << 7) | col] = t;
    }
  }
  __syncthreads();
  for (int s = 1; s <= 7; ++s) {
    const int half = 1 << (s - 1);
    for (int idx = tid; idx < 8192; idx += nth) {
      const int col = idx & 127, j = idx >> 7;
      const int grp = j >> (s - 1), pos = j & (half - 1);
      const int i0 = (((grp << s) + pos) << 7) + col;
      const int i1 = i0 + (half << 7);
      const float ang = dir * 3.14159265358979f * (float)pos / (float)half;
      float sn, cs;
      __sincosf(ang, &sn, &cs);
      const float2 u = c[i0], v = c[i1];
      const float tr = cs * v.x - sn * v.y;
      const float ti = cs * v.y + sn * v.x;
      c[i0] = make_float2(u.x + tr, u.y + ti);
      c[i1] = make_float2(u.x - tr, u.y - ti);
    }
    __syncthreads();
  }
}

// ---------------------------------------------------------------------------
// Per-image: bilinear scatter-splat (LDS atomics) -> 5x5 gaussian -> FFT
// -> CTF multiply (Hermitian extension of half-spectrum) -> inverse FFT.
// Dynamic LDS: 64KB accumulator + 128KB complex grid = 192KB (< 320KB/WGP).
// ---------------------------------------------------------------------------
__global__ void k_render(const float* __restrict__ params, const float* __restrict__ coords,
                         const float* __restrict__ values, const float* __restrict__ ctf,
                         float* __restrict__ out) {
  extern __shared__ float smem[];
  float*  accum = smem;                    // NPIX floats
  float2* csm   = (float2*)(smem + NPIX);  // NPIX float2
  const int b = blockIdx.x, tid = threadIdx.x, nth = blockDim.x;

  const float* pp = params + b * 8;
  const float r00 = pp[0], r01 = pp[1], r02 = pp[2];
  const float r10 = pp[3], r11 = pp[4], r12 = pp[5];
  const float zs0 = pp[6], zs1 = pp[7];

  for (int i = tid; i < NPIX; i += nth) accum[i] = 0.f;
  __syncthreads();

  // bilinear scatter-add of 65536 projected points
  for (int p = tid; p < NPTS; p += nth) {
    const float c0 = coords[p * 3 + 0], c1 = coords[p * 3 + 1], c2 = coords[p * 3 + 2];
    const float px = c0 * r00 + c1 * r01 + c2 * r02 + zs0 + 64.0f;
    const float py = c0 * r10 + c1 * r11 + c2 * r12 + zs1 + 64.0f;
    const float fx = floorf(px), fy = floorf(py);
    const int ix = (int)fx, iy = (int)fy;
    const float dx = px - fx, dy = py - fy;
    const float w = values[p];
    const int x0 = min(max(ix, 0), 127),     x1 = min(max(ix + 1, 0), 127);
    const int y0 = min(max(iy, 0), 127),     y1 = min(max(iy + 1, 0), 127);
    atomicAdd(&accum[(y0 << 7) + x0], w * (1.f - dx) * (1.f - dy));
    atomicAdd(&accum[(y0 << 7) + x1], w * dx * (1.f - dy));
    atomicAdd(&accum[(y1 << 7) + x0], w * (1.f - dx) * dy);
    atomicAdd(&accum[(y1 << 7) + x1], w * dx * dy);
  }
  __syncthreads();

  // 5x5 gaussian (sigma=1), 'SAME' zero padding -> real part of complex grid
  const float g0 = expf(-2.0f), g1 = expf(-0.5f);
  const float gs = g0 + g1 + 1.0f + g1 + g0;
  const float invn = 1.0f / (gs * gs);
  const float gg[5] = {g0, g1, 1.0f, g1, g0};
  for (int i = tid; i < NPIX; i += nth) {
    const int y = i >> 7, x = i & 127;
    float s = 0.f;
    #pragma unroll
    for (int dy2 = -2; dy2 <= 2; ++dy2) {
      const int yy = y + dy2;
      if (yy < 0 || yy > 127) continue;
      #pragma unroll
      for (int dx2 = -2; dx2 <= 2; ++dx2) {
        const int xx = x + dx2;
        if (xx < 0 || xx > 127) continue;
        s += accum[(yy << 7) + xx] * gg[dy2 + 2] * gg[dx2 + 2];
      }
    }
    csm[i] = make_float2(s * invn, 0.f);
  }

  // forward 2D FFT
  fft128_rows(csm, -1.0f, tid, nth);
  fft128_cols(csm, -1.0f, tid, nth);

  // multiply full spectrum by real CTF (Hermitian extension for kx > 64)
  const float* cb = ctf + (size_t)b * XSZ * 65;
  __syncthreads();
  for (int i = tid; i < NPIX; i += nth) {
    const int ky = i >> 7, kx = i & 127;
    const float cv = (kx <= 64) ? cb[ky * 65 + kx]
                                : cb[((XSZ - ky) & 127) * 65 + (XSZ - kx)];
    csm[i].x *= cv;
    csm[i].y *= cv;
  }

  // inverse 2D FFT + 1/N^2 scaling, write real part
  fft128_cols(csm, 1.0f, tid, nth);
  fft128_rows(csm, 1.0f, tid, nth);
  const float sc = 1.0f / (float)NPIX;
  for (int i = tid; i < NPIX; i += nth) out[(size_t)b * NPIX + i] = csm[i].x * sc;
}

// ---------------------------------------------------------------------------
extern "C" void kernel_launch(void* const* d_in, const int* in_sizes, int n_in,
                              void* d_out, int out_size, void* d_ws, size_t ws_size,
                              hipStream_t stream) {
  const float* images  = (const float*)d_in[0];
  const float* eps_a   = (const float*)d_in[1];
  const float* eps_s   = (const float*)d_in[2];
  const float* W0      = (const float*)d_in[3];
  const float* b0      = (const float*)d_in[4];
  const float* W_hid   = (const float*)d_in[5];
  const float* b_hid   = (const float*)d_in[6];
  const float* Wy      = (const float*)d_in[7];
  const float* by      = (const float*)d_in[8];
  const float* Wz      = (const float*)d_in[9];
  const float* bz      = (const float*)d_in[10];
  const float* bn_y    = (const float*)d_in[11];
  const float* bn_z    = (const float*)d_in[12];
  const float* W_algn  = (const float*)d_in[13];
  const float* b_algn  = (const float*)d_in[14];
  const float* W_shift = (const float*)d_in[15];
  const float* b_shift = (const float*)d_in[16];
  const float* coords  = (const float*)d_in[17];
  const float* values  = (const float*)d_in[18];
  const float* rotb    = (const float*)d_in[19];
  const float* tiltb   = (const float*)d_in[20];
  const float* psib    = (const float*)d_in[21];
  const float* ctf     = (const float*)d_in[22];
  float* out = (float*)d_out;

  float* ws     = (float*)d_ws;
  float* xstd   = ws;                                // 64*16384
  float* hA     = xstd + (size_t)BATCH * NPIX;       // 64*1024
  float* hB     = hA + (size_t)BATCH * HD;           // 64*1024
  float* ybuf   = hB + (size_t)BATCH * HD;           // 64*1024
  float* zbuf   = ybuf + (size_t)BATCH * HD;         // 64*1024
  float* params = zbuf + (size_t)BATCH * HD;         // 64*8

  k_standardize<<<BATCH, 256, 0, stream>>>(images, xstd);

  // encoder tower: one wave per 16-column tile -> 64 blocks per layer
  k_gemm_relu<<<HD / 16, 32, 0, stream>>>(xstd, W0, b0, nullptr, hA, NPIX, HD);
  float* cur = hA;
  float* nxt = hB;
  for (int i = 0; i < 11; ++i) {
    k_gemm_relu<<<HD / 16, 32, 0, stream>>>(cur, W_hid + (size_t)i * HD * HD,
                                            b_hid + (size_t)i * HD, nullptr, nxt, HD, HD);
    float* t = cur; cur = nxt; nxt = t;
  }
  k_gemm_relu<<<HD / 16, 32, 0, stream>>>(cur, Wy, by, bn_y, ybuf, HD, HD);
  k_gemm_relu<<<HD / 16, 32, 0, stream>>>(cur, Wz, bz, bn_z, zbuf, HD, HD);

  k_heads<<<BATCH, 256, 0, stream>>>(ybuf, zbuf, W_algn, b_algn, W_shift, b_shift,
                                     eps_a, eps_s, rotb, tiltb, psib, params);

  const size_t smem = (size_t)NPIX * sizeof(float) + (size_t)NPIX * sizeof(float2); // 192KB
  (void)hipFuncSetAttribute((const void*)k_render,
                            hipFuncAttributeMaxDynamicSharedMemorySize, (int)smem);
  k_render<<<BATCH, 256, smem, stream>>>(params, coords, values, ctf, out);
}
